// BERTGCNModel_68710886801893
// MI455X (gfx1250) — compile-verified, matrix operand order checked by reading, not run on hardware
//
#include <hip/hip_runtime.h>
#include <hip/hip_bf16.h>

// ---------------------------------------------------------------------------
// BERT-GCN forward for MI455X (gfx1250, wave32, WMMA).
// GEMMs run as bf16 WMMA (v_wmma_f32_16x16x32_bf16) with f32 accumulation.
// Each wave produces a 16x64 output strip (4 tiles share one A fragment) to
// raise WMMA-per-load density. Aggregation is gather/scale/scatter with
// global f32 atomics (L2-resident on the 192MB L2).
// ---------------------------------------------------------------------------

typedef __bf16 bf16_t;
typedef __attribute__((ext_vector_type(16))) __bf16 v16bf;
typedef __attribute__((ext_vector_type(8)))  __bf16 v8bf;
typedef __attribute__((ext_vector_type(8)))  float  v8f;

#define TPB 256

// ---------------- simple utility kernels ----------------

__global__ void k_fill_f32(float* __restrict__ p, float v, long long n) {
    long long i = (long long)blockIdx.x * blockDim.x + threadIdx.x;
    if (i < n) p[i] = v;
}

__global__ void k_deg_edges(const long long* __restrict__ ei,
                            const float* __restrict__ ew,
                            float* __restrict__ deg, int E) {
    int e = blockIdx.x * blockDim.x + threadIdx.x;
    if (e < E) {
        int c = (int)ei[(long long)E + e];   // col = target
        atomicAdd(&deg[c], ew[e]);
    }
}

__global__ void k_rsqrt_inplace(float* __restrict__ d, int n) {
    int i = blockIdx.x * blockDim.x + threadIdx.x;
    if (i < n) {
        float v = d[i];
        d[i] = (v > 0.0f) ? rsqrtf(v) : 0.0f;
    }
}

__global__ void k_f32_to_bf16(const float* __restrict__ s,
                              bf16_t* __restrict__ d, long long n) {
    long long i = (long long)blockIdx.x * blockDim.x + threadIdx.x;
    if (i < n) d[i] = (bf16_t)s[i];
}

// Wt[m*K + k] = W[k*M + m], converted to bf16 (so WMMA B loads are contiguous)
__global__ void k_transpose_to_bf16(const float* __restrict__ W,
                                    bf16_t* __restrict__ Wt, int K, int M) {
    int i = blockIdx.x * blockDim.x + threadIdx.x;
    if (i < K * M) {
        int m = i / K;
        int k = i - m * K;
        Wt[i] = (bf16_t)W[(long long)k * M + m];
    }
}

// ---------------- WMMA GEMM: C[N,M] = A[N,K] * Bt[M,K]^T ----------------
// 8 waves per block; each wave computes a 16x64 strip (four 16x16 tiles that
// share one A fragment), K-loop step 32. Per-lane A/B fragment layout per
// CDNA5 ISA 7.12.2 (16-bit A-matrix 16x32):
//   lanes 0-15:  kb=0  -> elems 0..7 = K 0..7,  elems 8..15 = K 16..23
//   lanes 16-31: kb=8  -> elems 0..7 = K 8..15, elems 8..15 = K 24..31
// Requires M % 64 == 0 (true: 512, 128).
__global__ void k_gemm_wmma_bf16(const bf16_t* __restrict__ A,
                                 const bf16_t* __restrict__ Bt,
                                 float* __restrict__ C,
                                 int nrows, int K, int M) {
    const int lane = threadIdx.x & 31;
    const int wave = threadIdx.x >> 5;
    const int m0 = (blockIdx.x * 8 + wave) * 16;   // output row tile
    const int n0 = blockIdx.y * 64;                // output col strip
    if (m0 >= nrows) return;                       // uniform per wave

    const int r  = lane & 15;
    const int kb = (lane < 16) ? 0 : 8;

    int arow = m0 + r;
    if (arow >= nrows) arow = nrows - 1;           // clamp (EXEC must stay full)
    const bf16_t* __restrict__ ap = A  + (long long)arow * K + kb;
    const bf16_t* __restrict__ bp = Bt + (long long)(n0 + r) * K + kb;
    const int bstride = 16 * K;                    // 16 rows of Bt per tile

    v8f acc0 = {}, acc1 = {}, acc2 = {}, acc3 = {};
    for (int k = 0; k < K; k += 32) {
        union { v16bf v; v8bf h[2]; } a, b0, b1, b2, b3;
        a.h[0]  = *(const v8bf*)(ap + k);                     // global_load_b128
        a.h[1]  = *(const v8bf*)(ap + k + 16);
        const bf16_t* bk = bp + k;
        b0.h[0] = *(const v8bf*)(bk);
        b0.h[1] = *(const v8bf*)(bk + 16);
        b1.h[0] = *(const v8bf*)(bk + bstride);
        b1.h[1] = *(const v8bf*)(bk + bstride + 16);
        b2.h[0] = *(const v8bf*)(bk + 2 * bstride);
        b2.h[1] = *(const v8bf*)(bk + 2 * bstride + 16);
        b3.h[0] = *(const v8bf*)(bk + 3 * bstride);
        b3.h[1] = *(const v8bf*)(bk + 3 * bstride + 16);
        acc0 = __builtin_amdgcn_wmma_f32_16x16x32_bf16(false, a.v, false, b0.v,
                                                       (short)0, acc0, false, false);
        acc1 = __builtin_amdgcn_wmma_f32_16x16x32_bf16(false, a.v, false, b1.v,
                                                       (short)0, acc1, false, false);
        acc2 = __builtin_amdgcn_wmma_f32_16x16x32_bf16(false, a.v, false, b2.v,
                                                       (short)0, acc2, false, false);
        acc3 = __builtin_amdgcn_wmma_f32_16x16x32_bf16(false, a.v, false, b3.v,
                                                       (short)0, acc3, false, false);
    }

    // C/D layout: VGPR i -> M = i (lanes 0-15) or i+8 (lanes 16-31), N = lane&15
    const int col   = n0 + (lane & 15);
    const int rbase = m0 + ((lane >> 4) << 3);
    v8f accs[4] = {acc0, acc1, acc2, acc3};
#pragma unroll
    for (int t = 0; t < 4; ++t) {
#pragma unroll
        for (int i = 0; i < 8; ++i) {
            int rr = rbase + i;
            if (rr < nrows) C[(long long)rr * M + col + t * 16] = accs[t][i];
        }
    }
}

// ---------------- edge aggregation: out[col] += H[row] * norm ----------------
__global__ void k_agg_edges(const long long* __restrict__ ei,
                            const float* __restrict__ ew,
                            const float* __restrict__ dinv,
                            const float* __restrict__ H,
                            float* __restrict__ out, int E, int F) {
    const int chunks = F >> 2;
    long long idx = (long long)blockIdx.x * blockDim.x + threadIdx.x;
    if (idx >= (long long)E * chunks) return;
    int e = (int)(idx / chunks);
    int f = (int)(idx - (long long)e * chunks) << 2;
    int rs = (int)ei[e];
    int ct = (int)ei[(long long)E + e];
    float norm = dinv[rs] * ew[e] * dinv[ct];
    const float4 h = *(const float4*)(H + (long long)rs * F + f);
    float* o = out + (long long)ct * F + f;
    atomicAdd(o + 0, h.x * norm);
    atomicAdd(o + 1, h.y * norm);
    atomicAdd(o + 2, h.z * norm);
    atomicAdd(o + 3, h.w * norm);
}

// ---------------- finalize: relu(agg + XW*dinv^2 + bias) ----------------
__global__ void k_finalize(const float* __restrict__ agg,
                           const float* __restrict__ XW,
                           const float* __restrict__ dinv,
                           const float* __restrict__ bias,
                           float* __restrict__ out_f32,
                           bf16_t* __restrict__ out_bf16,
                           int n, int F) {
    long long idx = (long long)blockIdx.x * blockDim.x + threadIdx.x;
    if (idx >= (long long)n * F) return;
    int node = (int)(idx / F);
    int f    = (int)(idx - (long long)node * F);
    float di = dinv[node];
    float v  = agg[idx] + XW[idx] * di * di + bias[f];  // self-loop norm = dinv^2
    v = (v > 0.0f) ? v : 0.0f;
    if (out_f32)  out_f32[idx]  = v;
    if (out_bf16) out_bf16[idx] = (bf16_t)v;
}

// ---------------- FC head + sigmoid ----------------
__global__ void k_fc_sigmoid(const float* __restrict__ h,
                             const float* __restrict__ W,
                             const float* __restrict__ b,
                             float* __restrict__ out, int n, int K, int C) {
    long long idx = (long long)blockIdx.x * blockDim.x + threadIdx.x;
    if (idx >= (long long)n * C) return;
    int node = (int)(idx / C);
    int c    = (int)(idx - (long long)node * C);
    const float* hp = h + (long long)node * K;
    float s = b[c];
#pragma unroll 8
    for (int k = 0; k < K; ++k) s = fmaf(hp[k], W[k * C + c], s);
    out[idx] = 1.0f / (1.0f + __expf(-s));
}

// ---------------------------------------------------------------------------

extern "C" void kernel_launch(void* const* d_in, const int* in_sizes, int n_in,
                              void* d_out, int out_size, void* d_ws, size_t ws_size,
                              hipStream_t stream) {
    const float*     x   = (const float*)d_in[0];       // [N,1024]
    const long long* ei  = (const long long*)d_in[1];   // [2,E] int64
    const float*     ew  = (const float*)d_in[2];       // [E]
    const float*     W1  = (const float*)d_in[3];       // [1024,512]
    const float*     b1  = (const float*)d_in[4];
    const float*     W2  = (const float*)d_in[5];       // [512,128]
    const float*     b2  = (const float*)d_in[6];
    const float*     Wfc = (const float*)d_in[7];       // [128,14]
    const float*     bfc = (const float*)d_in[8];
    float* out = (float*)d_out;

    const int D0 = 1024, D1 = 512, D2 = 128, NC = 14;
    const int N = in_sizes[0] / D0;
    const int E = in_sizes[1] / 2;

    // ---- workspace layout (aliased; peak ~310 MB) ----
    char*  ws  = (char*)d_ws;
    size_t off = 0;
    auto alloc = [&](size_t bytes) -> void* {
        void* p = ws + off;
        off = (off + bytes + 255) & ~(size_t)255;
        return p;
    };
    float*  deg  = (float*)alloc((size_t)N * 4);            // -> dinv in place
    bf16_t* xb   = (bf16_t*)alloc((size_t)N * D0 * 2);      // X in bf16
    bf16_t* w1t  = (bf16_t*)alloc((size_t)D0 * D1 * 2);     // W1^T bf16
    bf16_t* w2t  = (bf16_t*)alloc((size_t)D1 * D2 * 2);     // W2^T bf16
    float*  xw1  = (float*)alloc((size_t)N * D1 * 4);       // X@W1
    float*  agg1 = (float*)alloc((size_t)N * D1 * 4);       // layer-1 aggregation
    // aliases (lifetimes do not overlap):
    bf16_t* h1b  = xb;                                      // relu(h1) bf16 over xb
    float*  xw2  = xw1;                                     // H1@W2 over xw1
    float*  agg2 = agg1;                                    // layer-2 agg over agg1
    float*  h2   = (float*)((char*)agg1 +
                     (((size_t)N * D2 * 4 + 255) & ~(size_t)255)); // after agg2

    const int T = TPB;
    auto blocks1d = [](long long n, int t) { return (unsigned)((n + t - 1) / t); };

    // ---- degrees / normalization ----
    k_fill_f32<<<blocks1d(N, T), T, 0, stream>>>(deg, 1.0f, N);   // self-loop wt
    k_deg_edges<<<blocks1d(E, T), T, 0, stream>>>(ei, ew, deg, E);
    k_rsqrt_inplace<<<blocks1d(N, T), T, 0, stream>>>(deg, N);    // deg -> dinv

    // ---- precision conversion / weight transpose ----
    k_f32_to_bf16<<<blocks1d((long long)N * D0, T), T, 0, stream>>>(x, xb, (long long)N * D0);
    k_transpose_to_bf16<<<blocks1d((long long)D0 * D1, T), T, 0, stream>>>(W1, w1t, D0, D1);
    k_transpose_to_bf16<<<blocks1d((long long)D1 * D2, T), T, 0, stream>>>(W2, w2t, D1, D2);

    const unsigned row_tiles   = (unsigned)((N + 15) / 16);
    const unsigned gemm_blocks = (row_tiles + 7) / 8;

    // ---- layer 1 ----
    {
        dim3 g(gemm_blocks, D1 / 64);
        k_gemm_wmma_bf16<<<g, T, 0, stream>>>(xb, w1t, xw1, N, D0, D1);
    }
    k_fill_f32<<<blocks1d((long long)N * D1, T), T, 0, stream>>>(agg1, 0.0f, (long long)N * D1);
    k_agg_edges<<<blocks1d((long long)E * (D1 / 4), T), T, 0, stream>>>(ei, ew, deg, xw1, agg1, E, D1);
    k_finalize<<<blocks1d((long long)N * D1, T), T, 0, stream>>>(agg1, xw1, deg, b1,
                                                                 nullptr, h1b, N, D1);

    // ---- layer 2 ----
    {
        dim3 g(gemm_blocks, D2 / 64);
        k_gemm_wmma_bf16<<<g, T, 0, stream>>>(h1b, w2t, xw2, N, D1, D2);
    }
    k_fill_f32<<<blocks1d((long long)N * D2, T), T, 0, stream>>>(agg2, 0.0f, (long long)N * D2);
    k_agg_edges<<<blocks1d((long long)E * (D2 / 4), T), T, 0, stream>>>(ei, ew, deg, xw2, agg2, E, D2);
    k_finalize<<<blocks1d((long long)N * D2, T), T, 0, stream>>>(agg2, xw2, deg, b2,
                                                                 h2, nullptr, N, D2);

    // ---- FC head + sigmoid ----
    k_fc_sigmoid<<<blocks1d((long long)N * NC, T), T, 0, stream>>>(h2, Wfc, bfc, out, N, D2, NC);
}
